// OrcaLlamaDecoderLayer_74466142978538
// MI455X (gfx1250) — matrix operand order repaired and checked by reading, not compiled, not run on hardware
//
#include <hip/hip_runtime.h>
#include <hip/hip_bf16.h>
#include <stdint.h>

// ---------------------------------------------------------------------------
// OrcaLlamaDecoderLayer for MI455X (gfx1250), wave32 + WMMA bf16 16x16x32.
// GEMMs and attention stage tiles into LDS with CDNA5 async global->LDS copies
// (ASYNCcnt-tracked, double-buffered) so HBM streaming overlaps the WMMA pipe.
// ---------------------------------------------------------------------------

typedef __attribute__((ext_vector_type(16))) __bf16 v16bf;
typedef __attribute__((ext_vector_type(8)))  float  v8f;
typedef int v4i __attribute__((vector_size(4 * sizeof(int))));

#define D_MODEL 2048
#define NHEAD   16
#define HDIM    128
#define FDIM    8192
#define TOTAL   9216

#if defined(__gfx1250__)
#define USE_ASYNC_LDS 1
#else
#define USE_ASYNC_LDS 0
#endif

union FragB {          // one WMMA 16-bit A/B fragment (8 VGPRs / 32 bytes)
  uint4          q[2];
  unsigned short s[16];
  v16bf          v;
};

__device__ __forceinline__ unsigned short f2bf(float f) {
  union { float f; unsigned u; } v; v.f = f;
  unsigned r = v.u + 0x7FFFu + ((v.u >> 16) & 1u);   // round-to-nearest-even
  return (unsigned short)(r >> 16);
}

__device__ __forceinline__ v8f v8f_zero() {
  v8f z = {0.f, 0.f, 0.f, 0.f, 0.f, 0.f, 0.f, 0.f};
  return z;
}

// ------------------- async global->LDS staging (16B/lane) -------------------
typedef __attribute__((address_space(1))) v4i glb_v4i;
typedef __attribute__((address_space(3))) v4i lds_v4i;

__device__ __forceinline__ void stage16(const unsigned short* g,
                                        unsigned short* l) {
#if USE_ASYNC_LDS
#if __has_builtin(__builtin_amdgcn_global_load_async_to_lds_b128)
  __builtin_amdgcn_global_load_async_to_lds_b128(
      (glb_v4i*)(uintptr_t)g, (lds_v4i*)(uintptr_t)l, 0, 0);
#else
  // GV-mode async load: vdst = LDS byte address, vaddr = 64-bit global addr.
  unsigned lofs = (unsigned)(uintptr_t)(lds_v4i*)(uintptr_t)l;
  asm volatile("global_load_async_to_lds_b128 %0, %1, off"
               :: "v"(lofs), "v"(g) : "memory");
#endif
#else
  *(uint4*)l = *(const uint4*)g;
#endif
}

template <int N>
__device__ __forceinline__ void stage_wait() {
#if USE_ASYNC_LDS
#if __has_builtin(__builtin_amdgcn_s_wait_asynccnt)
  __builtin_amdgcn_s_wait_asynccnt(N);
#else
  asm volatile("s_wait_asynccnt %0" :: "i"(N) : "memory");
#endif
#endif
}

// --------------------------- fp32 -> bf16 convert ---------------------------
__global__ void f32_to_bf16_kernel(const float* __restrict__ in,
                                   unsigned short* __restrict__ out, int n) {
  int i = blockIdx.x * blockDim.x + threadIdx.x;
  int stride = gridDim.x * blockDim.x;
  for (; i < n; i += stride) out[i] = f2bf(in[i]);
}

// ------------------------------- RMSNorm ------------------------------------
__global__ __launch_bounds__(256)
void rmsnorm_bf16_kernel(const float* __restrict__ x, const float* __restrict__ w,
                         unsigned short* __restrict__ out) {
  __shared__ float red[256];
  const int row = blockIdx.x;
  const float* xr = x + (size_t)row * D_MODEL;
  float ss = 0.f;
  for (int c = threadIdx.x; c < D_MODEL; c += 256) { float v = xr[c]; ss += v * v; }
  red[threadIdx.x] = ss;
  __syncthreads();
  for (int s = 128; s > 0; s >>= 1) {
    if ((int)threadIdx.x < s) red[threadIdx.x] += red[threadIdx.x + s];
    __syncthreads();
  }
  const float rstd = rsqrtf(red[0] * (1.0f / D_MODEL) + 1e-6f);
  for (int c = threadIdx.x; c < D_MODEL; c += 256)
    out[(size_t)row * D_MODEL + c] = f2bf(xr[c] * rstd * w[c]);
}

// ------------------------ tiled WMMA GEMM: C = A * W^T ----------------------
// A: [M,K] bf16 row-major, W: [N,K] bf16 row-major (so W^T is K-major = B).
// Block tile 128x64, 8 waves (4 along M x 2 along N), each wave 2x2 WMMA tiles.
// Double-buffered async LDS staging: tile k+1 streams while tile k computes.
// EPI 0: out bf16 = acc + bias                    (QKV projection)
// EPI 1: out f32  = acc + bias? + resid?          (out-proj / gate / down)
// EPI 2: out bf16 = sigmoid(gatebuf) * acc        (fused gated-up)
template <int EPI>
__global__ __launch_bounds__(256)
void gemm_wmma_kernel(const unsigned short* __restrict__ A,
                      const unsigned short* __restrict__ W,
                      const float* __restrict__ bias,
                      const float* __restrict__ resid,
                      const float* __restrict__ gatebuf,
                      void* __restrict__ outp,
                      int M, int N, int K) {
  __shared__ unsigned short sA[2][128][32];
  __shared__ unsigned short sB[2][64][32];

  const int tid  = threadIdx.x;
  const int lane = tid & 31;
  const int wave = tid >> 5;
  const int wy   = wave & 3;                 // 4 waves along M
  const int wx   = wave >> 2;                // 2 waves along N
  const int mBase = blockIdx.y * 128;
  const int nBase = blockIdx.x * 64;
  const int half  = (lane < 16) ? 0 : 1;
  const int l16   = lane & 15;

  // Per-thread staging coordinates (3 async 16B moves per thread per k-tile).
  const int ar0 = tid >> 2,          ac0 = (tid & 3) * 8;        // A rows 0-63
  const int ar1 = (tid + 256) >> 2,  ac1 = ac0;                  // A rows 64-127
  const int br  = tid >> 2,          bc  = (tid & 3) * 8;        // B rows 0-63

  auto stageTile = [&](int kt, int buf) {
    const int k0 = kt * 32;
    stage16(&A[(size_t)(mBase + ar0) * K + k0 + ac0], &sA[buf][ar0][ac0]);
    stage16(&A[(size_t)(mBase + ar1) * K + k0 + ac1], &sA[buf][ar1][ac1]);
    stage16(&W[(size_t)(nBase + br)  * K + k0 + bc],  &sB[buf][br][bc]);
  };

  v8f acc[2][2];
  #pragma unroll
  for (int i = 0; i < 2; i++)
    #pragma unroll
    for (int j = 0; j < 2; j++) acc[i][j] = v8f_zero();

  const int NK = K >> 5;           // K / 32
  stageTile(0, 0);

  for (int kt = 0; kt < NK; ++kt) {
    const int cur = kt & 1;
    if (kt + 1 < NK) {
      stageTile(kt + 1, cur ^ 1);
      stage_wait<3>();             // async retires in order: tile kt resident
    } else {
      stage_wait<0>();
    }
    if (kt + 2 < NK) {             // warm L2 for the tile after next
      const int k2 = (kt + 2) * 32;
      __builtin_prefetch(&A[(size_t)(mBase + ar0) * K + k2 + ac0], 0, 1);
      __builtin_prefetch(&W[(size_t)(nBase + br)  * K + k2 + bc],  0, 1);
    }
    __syncthreads();

    FragB a[2], b[2];
    #pragma unroll
    for (int im = 0; im < 2; im++) {      // A layout: lane<16 -> K0-7/16-23
      int m = wy * 32 + im * 16 + l16;
      int kb = half * 8;
      a[im].q[0] = *(const uint4*)&sA[cur][m][kb];
      a[im].q[1] = *(const uint4*)&sA[cur][m][16 + kb];
    }
    #pragma unroll
    for (int in = 0; in < 2; in++) {      // B layout: lane<16 -> K0-7/16-23
      int n = wx * 32 + in * 16 + l16;
      int kb = half * 8;
      b[in].q[0] = *(const uint4*)&sB[cur][n][kb];
      b[in].q[1] = *(const uint4*)&sB[cur][n][kb + 16];
    }
    #pragma unroll
    for (int im = 0; im < 2; im++)
      #pragma unroll
      for (int in = 0; in < 2; in++)
        acc[im][in] = __builtin_amdgcn_wmma_f32_16x16x32_bf16(
            false, a[im].v, false, b[in].v, (short)0, acc[im][in], false, false);

    __syncthreads();               // all reads done before buffer reuse
  }

  // Epilogue. C layout: lane holds col (l16), rows r + 8*half per VGPR r.
  #pragma unroll
  for (int im = 0; im < 2; im++) {
    #pragma unroll
    for (int in = 0; in < 2; in++) {
      const int col = nBase + wx * 32 + in * 16 + l16;
      const float bv = (EPI != 2 && bias != nullptr) ? bias[col] : 0.f;
      #pragma unroll
      for (int r = 0; r < 8; r++) {
        const int row = mBase + wy * 32 + im * 16 + r + half * 8;
        const size_t idx = (size_t)row * N + col;
        float v = acc[im][in][r] + bv;
        if (EPI == 0) {
          ((unsigned short*)outp)[idx] = f2bf(v);
        } else if (EPI == 1) {
          if (resid != nullptr) v += resid[idx];
          ((float*)outp)[idx] = v;
        } else {
          const float g = gatebuf[idx];
          const float s = 1.f / (1.f + __expf(-g));
          ((unsigned short*)outp)[idx] = f2bf(s * v);
        }
      }
    }
  }
}

// --------------------------- causal flash attention -------------------------
// qkv: [TOTAL, 3*D_MODEL] bf16, rows = [q | k | v], head h at col h*HDIM.
// One WG (128 threads / 4 waves) = 64 query rows of one (sequence, head);
// each wave owns one 16-row q tile. 32-key KV tiles double-buffered in LDS
// via async copies.
__global__ __launch_bounds__(128)
void flash_attn_kernel(const unsigned short* __restrict__ qkv,
                       unsigned short* __restrict__ attn_out) {
  // Static ragged layout from the reference (all lens are multiples of 64).
  const int SO[9] = {0, 512, 1536, 3584, 4352, 5888, 6784, 8064, 9216};
  const int CB[9] = {0,   8,   24,   56,   68,   92,  106,  126,  144};

  __shared__ unsigned short sK[2][32][128];
  __shared__ unsigned short sV[2][32][128];
  __shared__ unsigned short sP[4][16][32];

  const int tid  = threadIdx.x;
  const int lane = tid & 31;
  const int wave = tid >> 5;
  const int half = (lane < 16) ? 0 : 1;
  const int l16  = lane & 15;
  const int head = blockIdx.y;

  int qb = blockIdx.x;
  int s = 0;
  while (qb >= CB[s + 1]) s++;
  const int seq0 = SO[s];
  const int q0wg = (qb - CB[s]) * 64;       // WG's q base (local to sequence)
  const int q0   = q0wg + wave * 16;        // wave's q tile base
  const size_t ldq = 3 * D_MODEL;

  auto stageKV = [&](int kv0, int buf) {    // 8 async 16B moves per thread
    #pragma unroll
    for (int t = 0; t < 4; t++) {
      int i = tid + t * 128;
      int r = i >> 4, c = (i & 15) * 8;
      const unsigned short* src =
          qkv + (size_t)(seq0 + kv0 + r) * ldq + head * HDIM + c;
      stage16(src + D_MODEL,     &sK[buf][r][c]);
      stage16(src + 2 * D_MODEL, &sV[buf][r][c]);
    }
  };

  // Persistent Q fragments: 4 K-chunks covering HDIM = 128.
  FragB aQ[4];
  {
    const unsigned short* Qp =
        qkv + (size_t)(seq0 + q0 + l16) * ldq + head * HDIM;
    #pragma unroll
    for (int c = 0; c < 4; c++) {
      int kb = c * 32 + half * 8;
      aQ[c].q[0] = *(const uint4*)&Qp[kb];
      aQ[c].q[1] = *(const uint4*)&Qp[kb + 16];
    }
  }

  v8f acc[8];                               // 8 n-tiles x (8 rows per lane)
  #pragma unroll
  for (int i = 0; i < 8; i++) acc[i] = v8f_zero();
  float mrow[8], lrow[8];
  #pragma unroll
  for (int r = 0; r < 8; r++) { mrow[r] = -1e30f; lrow[r] = 0.f; }

  const float scale = 0.08838834764831845f;  // 1/sqrt(128)
  const int NKV = (q0wg + 64) >> 5;          // causal bound, 32-key tiles

  stageKV(0, 0);

  for (int it = 0; it < NKV; ++it) {
    const int cur = it & 1;
    const int kv0 = it * 32;
    if (it + 1 < NKV) {
      stageKV(kv0 + 32, cur ^ 1);
      stage_wait<8>();
    } else {
      stage_wait<0>();
    }
    __syncthreads();

    // Scores: two 16-key tiles, contraction over HDIM in 4 chunks of 32.
    v8f sc[2];
    #pragma unroll
    for (int t = 0; t < 2; t++) {
      sc[t] = v8f_zero();
      #pragma unroll
      for (int c = 0; c < 4; c++) {
        FragB bK;
        int key = t * 16 + l16;
        int kb = c * 32 + half * 8;
        bK.q[0] = *(const uint4*)&sK[cur][key][kb];
        bK.q[1] = *(const uint4*)&sK[cur][key][kb + 16];
        sc[t] = __builtin_amdgcn_wmma_f32_16x16x32_bf16(
            false, aQ[c].v, false, bK.v, (short)0, sc[t], false, false);
      }
    }

    // Online softmax. Rows 0-7 live in lanes 0-15, rows 8-15 in lanes 16-31
    // (WMMA C layout) -> xor-reduce with masks 1..8 stays within each half.
    float pr[2][8];
    #pragma unroll
    for (int r = 0; r < 8; r++) {
      const int qrow = q0 + r + half * 8;
      float s0 = sc[0][r] * scale;
      float s1 = sc[1][r] * scale;
      if (kv0 + l16 > qrow)      s0 = -1e30f;
      if (kv0 + 16 + l16 > qrow) s1 = -1e30f;
      float mx = fmaxf(s0, s1);
      #pragma unroll
      for (int d = 1; d < 16; d <<= 1) mx = fmaxf(mx, __shfl_xor(mx, d, 32));
      const float mnew = fmaxf(mrow[r], mx);
      const float resc = __expf(mrow[r] - mnew);
      const float p0 = __expf(s0 - mnew);
      const float p1 = __expf(s1 - mnew);
      float ps = p0 + p1;
      #pragma unroll
      for (int d = 1; d < 16; d <<= 1) ps += __shfl_xor(ps, d, 32);
      lrow[r] = lrow[r] * resc + ps;
      mrow[r] = mnew;
      pr[0][r] = p0;
      pr[1][r] = p1;
      #pragma unroll
      for (int nt = 0; nt < 8; nt++) acc[nt][r] *= resc;
    }

    // Transpose P (C layout -> A layout) through per-wave LDS scratch.
    #pragma unroll
    for (int r = 0; r < 8; r++) {
      sP[wave][r + half * 8][l16]      = f2bf(pr[0][r]);
      sP[wave][r + half * 8][16 + l16] = f2bf(pr[1][r]);
    }
    FragB aP;   // DS ops are in-order per wave: stores above precede loads.
    {
      int kb = half * 16;
      aP.q[0] = *(const uint4*)&sP[wave][l16][kb];
      aP.q[1] = *(const uint4*)&sP[wave][l16][kb + 8];
    }

    // PV: contraction over 32 kv positions, 8 n-tiles cover HDIM = 128.
    #pragma unroll
    for (int nt = 0; nt < 8; nt++) {
      FragB bV;
      #pragma unroll
      for (int j = 0; j < 16; j++)
        bV.s[j] = sV[cur][half * 16 + j][nt * 16 + l16];
      acc[nt] = __builtin_amdgcn_wmma_f32_16x16x32_bf16(
          false, aP.v, false, bV.v, (short)0, acc[nt], false, false);
    }

    __syncthreads();               // reads done before this buffer is restaged
  }

  // Normalize and store: attn_out[row, head*HDIM + nt*16 + l16] (bf16).
  #pragma unroll
  for (int nt = 0; nt < 8; nt++) {
    #pragma unroll
    for (int r = 0; r < 8; r++) {
      const int row = seq0 + q0 + r + half * 8;
      const float v = acc[nt][r] / lrow[r];
      attn_out[(size_t)row * D_MODEL + head * HDIM + nt * 16 + l16] = f2bf(v);
    }
  }
}

// ------------------------------- launcher -----------------------------------
extern "C" void kernel_launch(void* const* d_in, const int* in_sizes, int n_in,
                              void* d_out, int out_size, void* d_ws,
                              size_t ws_size, hipStream_t stream) {
  (void)in_sizes; (void)n_in; (void)out_size; (void)ws_size;

  const float* hidden = (const float*)d_in[0];
  const float* ln1_w  = (const float*)d_in[1];
  const float* ln2_w  = (const float*)d_in[2];
  const float* in_w   = (const float*)d_in[3];   // [3D, D]
  const float* in_b   = (const float*)d_in[4];   // [3D]
  const float* out_w  = (const float*)d_in[5];   // [D, D]
  const float* out_b  = (const float*)d_in[6];   // [D]
  const float* gate_w = (const float*)d_in[7];   // [F, D]
  const float* up_w   = (const float*)d_in[8];   // [F, D]
  const float* down_w = (const float*)d_in[9];   // [D, F]

  char* ws = (char*)d_ws;
  size_t off = 0;
  auto alloc = [&](size_t bytes) -> void* {
    void* p = ws + off;
    off += (bytes + 255) & ~(size_t)255;
    return p;
  };

  const size_t N_INW  = (size_t)3 * D_MODEL * D_MODEL;   // 12.58M
  const size_t N_OUTW = (size_t)D_MODEL * D_MODEL;       //  4.19M
  const size_t N_FFW  = (size_t)FDIM * D_MODEL;          // 16.78M
  const size_t N_X    = (size_t)TOTAL * D_MODEL;         // 18.87M
  const size_t N_QKV  = (size_t)TOTAL * 3 * D_MODEL;     // 56.62M
  const size_t N_H    = (size_t)TOTAL * FDIM;            // 75.50M

  unsigned short* w_in_bf   = (unsigned short*)alloc(N_INW  * 2);
  unsigned short* w_out_bf  = (unsigned short*)alloc(N_OUTW * 2);
  unsigned short* w_gate_bf = (unsigned short*)alloc(N_FFW  * 2);
  unsigned short* w_up_bf   = (unsigned short*)alloc(N_FFW  * 2);
  unsigned short* w_down_bf = (unsigned short*)alloc(N_FFW  * 2);
  unsigned short* x_bf      = (unsigned short*)alloc(N_X    * 2);
  unsigned short* qkv_bf    = (unsigned short*)alloc(N_QKV  * 2);
  unsigned short* att_bf    = (unsigned short*)alloc(N_X    * 2);
  float*          resid2    = (float*)         alloc(N_X    * 4);
  float*          gbuf      = (float*)         alloc(N_H    * 4);
  unsigned short* h_bf      = (unsigned short*)alloc(N_H    * 2);

  // 1) weights -> bf16
  f32_to_bf16_kernel<<<4096, 256, 0, stream>>>(in_w,   w_in_bf,   (int)N_INW);
  f32_to_bf16_kernel<<<4096, 256, 0, stream>>>(out_w,  w_out_bf,  (int)N_OUTW);
  f32_to_bf16_kernel<<<4096, 256, 0, stream>>>(gate_w, w_gate_bf, (int)N_FFW);
  f32_to_bf16_kernel<<<4096, 256, 0, stream>>>(up_w,   w_up_bf,   (int)N_FFW);
  f32_to_bf16_kernel<<<4096, 256, 0, stream>>>(down_w, w_down_bf, (int)N_FFW);

  // 2) x = rmsnorm(hidden) -> bf16
  rmsnorm_bf16_kernel<<<TOTAL, 256, 0, stream>>>(hidden, ln1_w, x_bf);

  // 3) qkv = x @ in_proj^T + b  (bf16 out)   M=9216 N=6144 K=2048
  gemm_wmma_kernel<0><<<dim3(6144 / 64, TOTAL / 128), 256, 0, stream>>>(
      x_bf, w_in_bf, in_b, nullptr, nullptr, qkv_bf, TOTAL, 3 * D_MODEL, D_MODEL);

  // 4) causal flash attention (144 q-blocks x 16 heads)
  flash_attn_kernel<<<dim3(144, NHEAD), 128, 0, stream>>>(qkv_bf, att_bf);

  // 5) resid2 = hidden + attn @ out_proj^T + b   (fp32)
  gemm_wmma_kernel<1><<<dim3(D_MODEL / 64, TOTAL / 128), 256, 0, stream>>>(
      att_bf, w_out_bf, out_b, hidden, nullptr, resid2, TOTAL, D_MODEL, D_MODEL);

  // 6) x2 = rmsnorm(resid2) -> bf16 (reuse x_bf)
  rmsnorm_bf16_kernel<<<TOTAL, 256, 0, stream>>>(resid2, ln2_w, x_bf);

  // 7) gbuf = x2 @ gate^T (fp32)   N=8192
  gemm_wmma_kernel<1><<<dim3(FDIM / 64, TOTAL / 128), 256, 0, stream>>>(
      x_bf, w_gate_bf, nullptr, nullptr, nullptr, gbuf, TOTAL, FDIM, D_MODEL);

  // 8) h = sigmoid(gbuf) * (x2 @ up^T) -> bf16
  gemm_wmma_kernel<2><<<dim3(FDIM / 64, TOTAL / 128), 256, 0, stream>>>(
      x_bf, w_up_bf, nullptr, nullptr, gbuf, h_bf, TOTAL, FDIM, D_MODEL);

  // 9) out = resid2 + h @ down^T (fp32)   K=8192
  gemm_wmma_kernel<1><<<dim3(D_MODEL / 64, TOTAL / 128), 256, 0, stream>>>(
      h_bf, w_down_bf, nullptr, resid2, nullptr, (float*)d_out, TOTAL, D_MODEL, FDIM);
}
